// NMF2D_15023795601643
// MI455X (gfx1250) — matrix operand order repaired and checked by reading, not compiled, not run on hardware
//
#include <hip/hip_runtime.h>
#include <hip/hip_bf16.h>

// ---- problem constants (from reference) ----
#define Bsz 16
#define Dd  512
#define Nn  4096
#define Rr  64
#define EPSV 1e-6f
#define STEPS 7

typedef __attribute__((ext_vector_type(16))) _Float16 v16h;
typedef __attribute__((ext_vector_type(8)))  _Float16 v8h;
typedef __attribute__((ext_vector_type(8)))  float    v8f;

#define WMMA(a,b,c) __builtin_amdgcn_wmma_f32_16x16x32_f16(false,(a),false,(b),(short)0,(c),false,false)

// ---- per-lane operand loaders (K-contiguous source rows) ----
// A 16x32 f16: lane half h holds K = h*8..h*8+7 (elems 0..7) and 16+h*8.. (elems 8..15)
__device__ __forceinline__ v16h ldA(const _Float16* p, int half) {
  union { v16h v; v8h h[2]; } u;
  u.h[0] = *(const v8h*)(p + half * 8);
  u.h[1] = *(const v8h*)(p + 16 + half * 8);
  return u.v;
}
__device__ __forceinline__ v16h ldAf(const float* p, int half) {
  v16h r;
#pragma unroll
  for (int i = 0; i < 8; ++i) r[i] = (_Float16)p[half * 8 + i];
#pragma unroll
  for (int i = 0; i < 8; ++i) r[8 + i] = (_Float16)p[16 + half * 8 + i];
  return r;
}
// B 32x16 f16: lane half h holds K = h*8..h*8+7 (elems 0..7) and 16+h*8.. (elems 8..15)
__device__ __forceinline__ v16h ldB(const _Float16* p, int half) {
  union { v16h v; v8h h[2]; } u;
  u.h[0] = *(const v8h*)(p + half * 8);
  u.h[1] = *(const v8h*)(p + 16 + half * 8);
  return u.v;
}
__device__ __forceinline__ v16h ldBf(const float* p, int half) {
  v16h r;
#pragma unroll
  for (int i = 0; i < 8; ++i) r[i] = (_Float16)p[half * 8 + i];
#pragma unroll
  for (int i = 0; i < 8; ++i) r[8 + i] = (_Float16)p[16 + half * 8 + i];
  return r;
}

// ---- K0: x [B,D,N] f32 -> xh [B,D,N] f16 and xhT [B,N,D] f16 ----
__global__ void k_cvt_x(const float* __restrict__ x, _Float16* __restrict__ xh,
                        _Float16* __restrict__ xhT) {
  long i = (long)blockIdx.x * blockDim.x + threadIdx.x;   // over B*D*N
  float v = x[i];
  xh[i] = (_Float16)v;
  long n = i % Nn;
  long bd = i / Nn;
  long d = bd % Dd;
  long b = bd / Dd;
  xhT[(b * Nn + n) * Dd + d] = (_Float16)v;
}

// ---- K0b: bases f32 -> basesf (f32 work copy), bh [B,D,R] f16, bTh [B,R,D] f16 ----
__global__ void k_cvt_b(const float* __restrict__ bin, float* __restrict__ basesf,
                        _Float16* __restrict__ bh, _Float16* __restrict__ bTh) {
  int i = blockIdx.x * blockDim.x + threadIdx.x;          // over B*D*R
  float v = bin[i];
  basesf[i] = v;
  bh[i] = (_Float16)v;
  int r = i % Rr;
  int bd = i / Rr;
  int d = bd % Dd;
  int b = bd / Dd;
  bTh[((long)b * Rr + r) * Dd + d] = (_Float16)v;
}

// ---- gram: G[b] = T[b] @ T[b]^T  (T: [B, Rr, K] f16, G: [B, Rr, Rr] f32) ----
// one WG (8 waves) per batch; 16 output tiles of 16x16, 2 per wave.
__global__ void k_gram(const _Float16* __restrict__ T, int K, float* __restrict__ G) {
  int b = blockIdx.x;
  const _Float16* Tb = T + (long)b * Rr * K;
  float* Gb = G + b * Rr * Rr;
  int lane = threadIdx.x & 31, w = threadIdx.x >> 5;
  int half = lane >> 4, l15 = lane & 15;
  int tm = w >> 1;
  int tn0 = (w & 1) * 2;
  const _Float16* arow = Tb + (long)(tm * 16 + l15) * K;
  const _Float16* brow0 = Tb + (long)((tn0 + 0) * 16 + l15) * K;
  const _Float16* brow1 = Tb + (long)((tn0 + 1) * 16 + l15) * K;
  v8f acc0 = {}; v8f acc1 = {};
  for (int k = 0; k < K; k += 32) {
    v16h a = ldA(arow + k, half);
    v16h b0 = ldB(brow0 + k, half);
    v16h b1 = ldB(brow1 + k, half);
    acc0 = WMMA(a, b0, acc0);
    acc1 = WMMA(a, b1, acc1);
  }
  int row = tm * 16 + half * 8;
#pragma unroll
  for (int v = 0; v < 8; ++v) {
    Gb[(row + v) * Rr + tn0 * 16 + l15] = acc0[v];
    Gb[(row + v) * Rr + (tn0 + 1) * 16 + l15] = acc1[v];
  }
}

// ---- coef kernel: num = xr^T @ bases (K=D), then softmax (mode 0) or
//      coef *= num / (coef@btb + eps) (mode 1). WG tile: 128 n-rows x 64 r. ----
__global__ void k_coef(const _Float16* __restrict__ xhT, const _Float16* __restrict__ bTh,
                       const float* __restrict__ btb, float* __restrict__ coef,
                       _Float16* __restrict__ cTh, int mode) {
  int b = blockIdx.y;
  int n0 = blockIdx.x * 128;
  int lane = threadIdx.x & 31, w = threadIdx.x >> 5;
  int half = lane >> 4, l15 = lane & 15;

  const _Float16* A = xhT + ((long)b * Nn + n0 + w * 16 + l15) * Dd;
  const _Float16* Bb = bTh + (long)b * Rr * Dd;
  v8f acc[4] = {};
  for (int k = 0; k < Dd; k += 32) {
    v16h a = ldA(A + k, half);
#pragma unroll
    for (int j = 0; j < 4; ++j) {
      v16h bm = ldB(Bb + (long)(j * 16 + l15) * Dd + k, half);
      acc[j] = WMMA(a, bm, acc[j]);
    }
  }

  _Float16* cThB = cTh + (long)b * Rr * Nn;
  int rowLoc = w * 16 + half * 8;

  if (mode == 0) {
    // softmax over r (64 values per n-row; row lives in 16 lanes of one half x 4 tiles)
#pragma unroll
    for (int v = 0; v < 8; ++v) {
      float m = fmaxf(fmaxf(acc[0][v], acc[1][v]), fmaxf(acc[2][v], acc[3][v]));
#pragma unroll
      for (int mk = 1; mk < 16; mk <<= 1) m = fmaxf(m, __shfl_xor(m, mk, 32));
      float e[4]; float s = 0.f;
#pragma unroll
      for (int j = 0; j < 4; ++j) { e[j] = __expf(acc[j][v] - m); s += e[j]; }
#pragma unroll
      for (int mk = 1; mk < 16; mk <<= 1) s += __shfl_xor(s, mk, 32);
      float inv = 1.0f / s;
      int n = n0 + rowLoc + v;
#pragma unroll
      for (int j = 0; j < 4; ++j) {
        float c = e[j] * inv;
        int r = j * 16 + l15;
        coef[((long)b * Nn + n) * Rr + r] = c;
        cThB[(long)r * Nn + n] = (_Float16)c;
      }
    }
  } else {
    // den = coef @ btb via WMMA (btb symmetric -> B rows K-contiguous)
    const float* btbB = btb + b * Rr * Rr;
    const float* Acf = coef + ((long)b * Nn + n0 + w * 16 + l15) * Rr;
    v8f den[4] = {};
#pragma unroll
    for (int k = 0; k < Rr; k += 32) {
      v16h a = ldAf(Acf + k, half);
#pragma unroll
      for (int j = 0; j < 4; ++j) {
        v16h bm = ldBf(btbB + (j * 16 + l15) * Rr + k, half);
        den[j] = WMMA(a, bm, den[j]);
      }
    }
#pragma unroll
    for (int v = 0; v < 8; ++v) {
      int n = n0 + rowLoc + v;
#pragma unroll
      for (int j = 0; j < 4; ++j) {
        int r = j * 16 + l15;
        long idx = ((long)b * Nn + n) * Rr + r;
        float c = coef[idx];
        float cn = c * acc[j][v] / (den[j][v] + EPSV);
        coef[idx] = cn;
        cThB[(long)r * Nn + n] = (_Float16)cn;
      }
    }
  }
}

// ---- bases kernel: num2^T = coef^T @ xr^T (K=N), den2^T = ctc @ bases^T (K=R),
//      bases *= num2/(den2+eps). WG tile: 64 r-rows x 128 d-cols. ----
__global__ void k_bases(const _Float16* __restrict__ xh, const _Float16* __restrict__ cTh,
                        const float* __restrict__ ctc, float* __restrict__ basesf,
                        _Float16* __restrict__ bh, _Float16* __restrict__ bTh) {
  int b = blockIdx.y;
  int d0 = blockIdx.x * 128;
  int lane = threadIdx.x & 31, w = threadIdx.x >> 5;
  int half = lane >> 4, l15 = lane & 15;

  const _Float16* Bx = xh + ((long)b * Dd + d0 + w * 16 + l15) * Nn; // col d, K=n contiguous
  const _Float16* cThB = cTh + (long)b * Rr * Nn;
  v8f acc[4] = {};
  for (int k = 0; k < Nn; k += 32) {
    v16h bm = ldB(Bx + k, half);
#pragma unroll
    for (int j = 0; j < 4; ++j) {
      v16h a = ldA(cThB + (long)(j * 16 + l15) * Nn + k, half);
      acc[j] = WMMA(a, bm, acc[j]);
    }
  }

  // den2^T: A = ctc (symmetric, K=r contiguous), B = basesf rows (d fixed, r contiguous)
  const float* ctcB = ctc + b * Rr * Rr;
  const float* Bbase = basesf + ((long)b * Dd + d0 + w * 16 + l15) * Rr;
  v8f den[4] = {};
#pragma unroll
  for (int k = 0; k < Rr; k += 32) {
    v16h bm = ldBf(Bbase + k, half);
#pragma unroll
    for (int j = 0; j < 4; ++j) {
      v16h a = ldAf(ctcB + (j * 16 + l15) * Rr + k, half);
      den[j] = WMMA(a, bm, den[j]);
    }
  }

  int d = d0 + w * 16 + l15;
#pragma unroll
  for (int v = 0; v < 8; ++v) {
#pragma unroll
    for (int j = 0; j < 4; ++j) {
      int r = j * 16 + v + 8 * half;
      long idx = ((long)b * Dd + d) * Rr + r;
      float bo = basesf[idx];
      float bn = bo * acc[j][v] / (den[j][v] + EPSV);
      basesf[idx] = bn;
      bh[idx] = (_Float16)bn;
      bTh[((long)b * Rr + r) * Dd + d] = (_Float16)bn;
    }
  }
}

// ---- out = bases @ coef^T : [B, D, N], K=R=64. WG tile: 128 d x 64 n. ----
__global__ void k_out(const _Float16* __restrict__ bh, const float* __restrict__ coef,
                      float* __restrict__ out) {
  int b = blockIdx.z;
  int d0 = blockIdx.y * 128;
  int n0 = blockIdx.x * 64;
  int lane = threadIdx.x & 31, w = threadIdx.x >> 5;
  int half = lane >> 4, l15 = lane & 15;

  const _Float16* A = bh + ((long)b * Dd + d0 + w * 16 + l15) * Rr;
  const float* coefB = coef + ((long)b * Nn + n0) * Rr;
  v8f acc[4] = {};
#pragma unroll
  for (int k = 0; k < Rr; k += 32) {
    v16h a = ldA(A + k, half);
#pragma unroll
    for (int j = 0; j < 4; ++j) {
      v16h bm = ldBf(coefB + (j * 16 + l15) * Rr + k, half);
      acc[j] = WMMA(a, bm, acc[j]);
    }
  }
#pragma unroll
  for (int v = 0; v < 8; ++v) {
    int d = d0 + w * 16 + v + 8 * half;
#pragma unroll
    for (int j = 0; j < 4; ++j) {
      out[((long)b * Dd + d) * Nn + n0 + j * 16 + l15] = acc[j][v];
    }
  }
}

extern "C" void kernel_launch(void* const* d_in, const int* in_sizes, int n_in,
                              void* d_out, int out_size, void* d_ws, size_t ws_size,
                              hipStream_t stream) {
  const float* x = (const float*)d_in[0];        // [B, D, H, W] = [B, D, N]
  const float* basesIn = (const float*)d_in[1];  // [B, D, R]
  float* out = (float*)d_out;                    // [B, D, N]

  // workspace carve-up (~156 MiB total; x copies stay L2-resident: 192 MB L2)
  char* p = (char*)d_ws;
  _Float16* xh  = (_Float16*)p; p += (size_t)Bsz * Dd * Nn * 2;  // 64 MiB
  _Float16* xhT = (_Float16*)p; p += (size_t)Bsz * Nn * Dd * 2;  // 64 MiB
  float*    coef   = (float*)p;    p += (size_t)Bsz * Nn * Rr * 4;  // 16 MiB
  _Float16* cTh    = (_Float16*)p; p += (size_t)Bsz * Rr * Nn * 2;  //  8 MiB
  float*    basesf = (float*)p;    p += (size_t)Bsz * Dd * Rr * 4;  //  2 MiB
  _Float16* bh     = (_Float16*)p; p += (size_t)Bsz * Dd * Rr * 2;  //  1 MiB
  _Float16* bTh    = (_Float16*)p; p += (size_t)Bsz * Rr * Dd * 2;  //  1 MiB
  float*    btb    = (float*)p;    p += (size_t)Bsz * Rr * Rr * 4;
  float*    ctc    = (float*)p;    p += (size_t)Bsz * Rr * Rr * 4;

  k_cvt_x<<<(Bsz * Dd * Nn) / 256, 256, 0, stream>>>(x, xh, xhT);
  k_cvt_b<<<(Bsz * Dd * Rr) / 256, 256, 0, stream>>>(basesIn, basesf, bh, bTh);

  dim3 gc(Nn / 128, Bsz);            // 32 x 16 WGs
  dim3 gb(Dd / 128, Bsz);            // 4 x 16 WGs
  k_coef<<<gc, 256, 0, stream>>>(xhT, bTh, btb, coef, cTh, 0);  // softmax init

  for (int s = 0; s < STEPS; ++s) {
    k_gram<<<Bsz, 256, 0, stream>>>(bTh, Dd, btb);              // btb = B^T B
    k_coef<<<gc, 256, 0, stream>>>(xhT, bTh, btb, coef, cTh, 1);
    k_gram<<<Bsz, 256, 0, stream>>>(cTh, Nn, ctc);              // ctc = C^T C
    k_bases<<<gb, 256, 0, stream>>>(xh, cTh, ctc, basesf, bh, bTh);
  }

  // final compute_coef
  k_gram<<<Bsz, 256, 0, stream>>>(bTh, Dd, btb);
  k_coef<<<gc, 256, 0, stream>>>(xhT, bTh, btb, coef, cTh, 1);

  // reconstruction
  k_out<<<dim3(Nn / 64, Dd / 128, Bsz), 256, 0, stream>>>(bh, coef, out);
}
